// AlibiCausalSelfAttention_80522046865832
// MI455X (gfx1250) — compile-verified, matrix operand order checked by reading, not run on hardware
//
#include <hip/hip_runtime.h>
#include <hip/hip_bf16.h>
#include <cstdint>

// MI455X / gfx1250, wave32. D = A*B + C via v_wmma_f32_16x16x32_bf16.
typedef __attribute__((ext_vector_type(16))) __bf16 v16bf;
typedef __attribute__((ext_vector_type(8)))  __bf16 v8bf;
typedef __attribute__((ext_vector_type(8)))  float  v8f;

// Async global->LDS copies (ASYNCcnt path), guarded: falls back to plain
// vector copies if this toolchain lacks the gfx1250 async builtins.
#if defined(__HIP_DEVICE_COMPILE__) && \
    __has_builtin(__builtin_amdgcn_global_load_async_to_lds_b128) && \
    __has_builtin(__builtin_amdgcn_s_wait_asynccnt)
#define USE_ASYNC_LDS 1
#else
#define USE_ASYNC_LDS 0
#endif

#if USE_ASYNC_LDS
#define GLOBAL_AS __attribute__((address_space(1)))
#define SHARED_AS __attribute__((address_space(3)))
typedef int vi4 __attribute__((vector_size(16)));   // matches builtin param
static __device__ __forceinline__ void async_cp16(const void* g, void* l) {
  __builtin_amdgcn_global_load_async_to_lds_b128(
      (GLOBAL_AS vi4*)g, (SHARED_AS vi4*)l, 0, 0);
}
static __device__ __forceinline__ void async_wait0() {
  __builtin_amdgcn_s_wait_asynccnt(0);
}
#else
static __device__ __forceinline__ void async_cp16(const void* g, void* l) {
  *(uint4*)l = *(const uint4*)g;
}
static __device__ __forceinline__ void async_wait0() {}
#endif

// 16-bit A/B fragment loader from a row-major LDS tile [row][stride].
// Per CDNA5 ISA 7.12.2: lane holds row (lane&15); lanes 0-15 take K chunks
// {0..7, 16..23}, lanes 16-31 take {8..15, 24..31}. Two 16B ds reads.
static __device__ __forceinline__ v16bf frag_ld(const __bf16* tile, int stride,
                                                int row, int kbase, int lane) {
  const int khalf = (lane < 16) ? 0 : 8;
  const __bf16* p = tile + row * stride + kbase + khalf;
  v8bf lo = *(const v8bf*)(p);
  v8bf hi = *(const v8bf*)(p + 16);
  return __builtin_shufflevector(lo, hi, 0,1,2,3,4,5,6,7,8,9,10,11,12,13,14,15);
}

static __device__ __forceinline__ v8f wmma_bf16(v16bf a, v16bf b, v8f c) {
  return __builtin_amdgcn_wmma_f32_16x16x32_bf16(false, a, false, b,
                                                 (short)0, c, false, false);
}

// ---------------------------------------------------------------------------
// Kernel 1: QKV GEMM.  [4096,1024] x [1024,3072] + bias -> bf16 Q/K/V [B,H,T,D]
// Block tile 128x128, 8 waves (4x2), wave tile 32x64 (2x4 WMMA), BK=32.
// ---------------------------------------------------------------------------
__global__ __launch_bounds__(256)
void qkv_gemm(const float* __restrict__ X, const float* __restrict__ W,
              const float* __restrict__ bias,
              __bf16* __restrict__ Qw, __bf16* __restrict__ Kw,
              __bf16* __restrict__ Vw)
{
  __shared__ __bf16 Asm[128 * 40];  // [row][k], stride 40 (bank-conflict pad)
  __shared__ __bf16 Bsm[128 * 40];  // B transposed: [n][k]
  const int m0 = blockIdx.x * 128;
  const int n0 = blockIdx.y * 128;
  const int tid = threadIdx.x;
  const int lane = tid & 31, wave = tid >> 5;
  const int wm = (wave & 3) * 32;
  const int wn = (wave >> 2) * 64;
  v8f acc[2][4] = {};

  const int ar = tid >> 3, ac = (tid & 7) * 4;        // A staging coords
  const int bn = (tid & 31) * 4, bk = tid >> 5;       // B staging coords

  for (int k0 = 0; k0 < 1024; k0 += 32) {
    __syncthreads();
    if (k0 + 32 < 1024) {   // prefetch next tiles (global_prefetch_b8)
      __builtin_prefetch(X + (size_t)(m0 + ar) * 1024 + k0 + 32 + ac, 0, 0);
      __builtin_prefetch(W + (size_t)(k0 + 32 + bk) * 3072 + n0 + bn, 0, 0);
    }
    {   // A tile 128x32 f32 -> bf16 LDS
      for (int p = 0; p < 4; ++p) {
        int row = ar + 32 * p;
        float4 f = *(const float4*)(X + (size_t)(m0 + row) * 1024 + k0 + ac);
        __bf16* d = &Asm[row * 40 + ac];
        d[0]=(__bf16)f.x; d[1]=(__bf16)f.y; d[2]=(__bf16)f.z; d[3]=(__bf16)f.w;
      }
    }
    {   // B tile 32x128 f32 -> transposed bf16 LDS
      for (int p = 0; p < 4; ++p) {
        int krow = bk + 8 * p;
        float4 f = *(const float4*)(W + (size_t)(k0 + krow) * 3072 + n0 + bn);
        Bsm[(bn+0)*40 + krow] = (__bf16)f.x;
        Bsm[(bn+1)*40 + krow] = (__bf16)f.y;
        Bsm[(bn+2)*40 + krow] = (__bf16)f.z;
        Bsm[(bn+3)*40 + krow] = (__bf16)f.w;
      }
    }
    __syncthreads();
    v16bf a0 = frag_ld(Asm, 40, wm +      (lane & 15), 0, lane);
    v16bf a1 = frag_ld(Asm, 40, wm + 16 + (lane & 15), 0, lane);
    for (int nt = 0; nt < 4; ++nt) {
      v16bf b = frag_ld(Bsm, 40, wn + nt * 16 + (lane & 15), 0, lane);
      acc[0][nt] = wmma_bf16(a0, b, acc[0][nt]);
      acc[1][nt] = wmma_bf16(a1, b, acc[1][nt]);
    }
  }

  // Epilogue: C layout lanes 0-15 -> M=r, lanes 16-31 -> M=r+8; N = lane&15.
  const int hi8 = (lane < 16) ? 0 : 8;
  for (int mt = 0; mt < 2; ++mt)
    for (int nt = 0; nt < 4; ++nt) {
      int n = n0 + wn + nt * 16 + (lane & 15);
      int which = n >> 10;            // 0=Q 1=K 2=V
      int c = n & 1023;
      int h = c >> 6, d = c & 63;
      float bv = bias[n];
      __bf16* dst = (which == 0) ? Qw : ((which == 1) ? Kw : Vw);
      for (int r = 0; r < 8; ++r) {
        int m = m0 + wm + mt * 16 + r + hi8;
        int bb = m >> 11, t = m & 2047;
        dst[(((size_t)(bb * 16 + h)) * 2048 + t) * 64 + d] =
            (__bf16)(acc[mt][nt][r] + bv);
      }
    }
}

// ---------------------------------------------------------------------------
// Kernel 2: flash attention with ALiBi + causal mask.
// grid (B*H=32, T/64=32), 128 threads = 4 waves; wave owns 16 query rows.
// ---------------------------------------------------------------------------
__global__ __launch_bounds__(128)
void attn_kernel(const __bf16* __restrict__ Qw, const __bf16* __restrict__ Kw,
                 const __bf16* __restrict__ Vw, float* __restrict__ Yw)
{
  __shared__ __bf16 Qt[64 * 72];      // [t][d]
  __shared__ __bf16 Kt[64 * 72];      // [s][d]
  __shared__ __bf16 Vt[64 * 72];      // transposed: [d][s]
  __shared__ __bf16 Pt[4][16 * 72];   // per-wave P tile [qrow][s]
  const int bh = blockIdx.x;
  const int qb = blockIdx.y;
  const int b = bh >> 4, h = bh & 15;
  const int tid = threadIdx.x, lane = tid & 31, wave = tid >> 5;
  const int hi8 = (lane < 16) ? 0 : 8;
  const float slope = exp2f(-(float)(h + 1));
  const float scale = 0.125f;                 // 1/sqrt(64)
  const size_t base = (size_t)bh * 2048 * 64;
  const int t0 = qb * 64;
  const int srow = tid >> 1, sd0 = (tid & 1) * 32;   // staging coords

  // stage Q tile (async if available; completion covered by first j-iter wait)
  {
    const __bf16* gs = Qw + base + (size_t)(t0 + srow) * 64 + sd0;
    __bf16* ls = &Qt[srow * 72 + sd0];
    for (int i = 0; i < 4; ++i) async_cp16(gs + i * 8, ls + i * 8);
  }

  float mrow[8], lrow[8];
  v8f o[4] = {};
  for (int r = 0; r < 8; ++r) { mrow[r] = -1e30f; lrow[r] = 0.f; }

  for (int j = 0; j <= qb; ++j) {
    __syncthreads();
    {   // stage K tile (async path -> GLOBAL_LOAD_ASYNC_TO_LDS_B128)
      const __bf16* gs = Kw + base + (size_t)(j * 64 + srow) * 64 + sd0;
      __bf16* ls = &Kt[srow * 72 + sd0];
      for (int i = 0; i < 4; ++i) async_cp16(gs + i * 8, ls + i * 8);
    }
    {   // stage V tile, transposed into Vt[d][s] (needs VALU, stays sync)
      const uint4* src = (const uint4*)(Vw + base + (size_t)(j*64 + srow)*64 + sd0);
      for (int i = 0; i < 4; ++i) {
        union { uint4 u; __bf16 e[8]; } tmp;
        tmp.u = src[i];
        for (int jj = 0; jj < 8; ++jj)
          Vt[(sd0 + i * 8 + jj) * 72 + srow] = tmp.e[jj];
      }
    }
    async_wait0();     // s_wait_asynccnt 0 (per-wave), then block barrier
    __syncthreads();

    // S = Q * K^T  (D=64 -> two K-steps of 32)
    v8f sAcc[4] = {};
    for (int ks = 0; ks < 2; ++ks) {
      v16bf aq = frag_ld(Qt, 72, wave * 16 + (lane & 15), ks * 32, lane);
      for (int nt = 0; nt < 4; ++nt) {
        v16bf bk = frag_ld(Kt, 72, nt * 16 + (lane & 15), ks * 32, lane);
        sAcc[nt] = wmma_bf16(aq, bk, sAcc[nt]);
      }
    }

    // online softmax; rows live per-VGPR, columns per-lane inside each half.
    const int keyc = j * 64 + (lane & 15);
    for (int r = 0; r < 8; ++r) {
      int tq = t0 + wave * 16 + r + hi8;
      float adj[4], rmax = -1e30f;
      for (int nt = 0; nt < 4; ++nt) {
        int key = keyc + nt * 16;
        float v = sAcc[nt][r] * scale + slope * (float)(key - tq);
        if (key > tq) v = -1e30f;           // causal
        adj[nt] = v;
        rmax = fmaxf(rmax, v);
      }
      for (int off = 8; off > 0; off >>= 1)
        rmax = fmaxf(rmax, __shfl_xor(rmax, off, 32));
      float mnew  = fmaxf(mrow[r], rmax);
      float alpha = __expf(mrow[r] - mnew);
      float sum = 0.f;
      for (int nt = 0; nt < 4; ++nt) {
        float p = __expf(adj[nt] - mnew);
        sum += p;
        Pt[wave][(r + hi8) * 72 + nt * 16 + (lane & 15)] = (__bf16)p;
      }
      for (int off = 8; off > 0; off >>= 1)
        sum += __shfl_xor(sum, off, 32);
      lrow[r] = lrow[r] * alpha + sum;
      mrow[r] = mnew;
      for (int dt = 0; dt < 4; ++dt) o[dt][r] *= alpha;
    }
    __syncthreads();   // P C-layout -> A-fragment layout via LDS

    // O += P * V   (s is the K-dim; V transposed so B frag reads are rows)
    for (int ks = 0; ks < 2; ++ks) {
      v16bf ap = frag_ld(Pt[wave], 72, lane & 15, ks * 32, lane);
      for (int dt = 0; dt < 4; ++dt) {
        v16bf bv = frag_ld(Vt, 72, dt * 16 + (lane & 15), ks * 32, lane);
        o[dt] = wmma_bf16(ap, bv, o[dt]);
      }
    }
  }

  // write y[B,T,C] f32
  for (int dt = 0; dt < 4; ++dt) {
    int d = dt * 16 + (lane & 15);
    for (int r = 0; r < 8; ++r) {
      int tq = t0 + wave * 16 + r + hi8;
      Yw[((size_t)(b * 2048 + tq)) * 1024 + h * 64 + d] = o[dt][r] / lrow[r];
    }
  }
}

// ---------------------------------------------------------------------------
// Kernel 3: output projection. [4096,1024] x [1024,1024] + bias -> f32 out
// Same structure as kernel 1 (block 128x128, wave 32x64).
// ---------------------------------------------------------------------------
__global__ __launch_bounds__(256)
void proj_gemm(const float* __restrict__ Y, const float* __restrict__ W,
               const float* __restrict__ bias, float* __restrict__ Out)
{
  __shared__ __bf16 Asm[128 * 40];
  __shared__ __bf16 Bsm[128 * 40];
  const int m0 = blockIdx.x * 128;
  const int n0 = blockIdx.y * 128;
  const int tid = threadIdx.x;
  const int lane = tid & 31, wave = tid >> 5;
  const int wm = (wave & 3) * 32;
  const int wn = (wave >> 2) * 64;
  v8f acc[2][4] = {};

  const int ar = tid >> 3, ac = (tid & 7) * 4;
  const int bn = (tid & 31) * 4, bk = tid >> 5;

  for (int k0 = 0; k0 < 1024; k0 += 32) {
    __syncthreads();
    if (k0 + 32 < 1024) {
      __builtin_prefetch(Y + (size_t)(m0 + ar) * 1024 + k0 + 32 + ac, 0, 0);
      __builtin_prefetch(W + (size_t)(k0 + 32 + bk) * 1024 + n0 + bn, 0, 0);
    }
    {
      for (int p = 0; p < 4; ++p) {
        int row = ar + 32 * p;
        float4 f = *(const float4*)(Y + (size_t)(m0 + row) * 1024 + k0 + ac);
        __bf16* d = &Asm[row * 40 + ac];
        d[0]=(__bf16)f.x; d[1]=(__bf16)f.y; d[2]=(__bf16)f.z; d[3]=(__bf16)f.w;
      }
    }
    {
      for (int p = 0; p < 4; ++p) {
        int krow = bk + 8 * p;
        float4 f = *(const float4*)(W + (size_t)(k0 + krow) * 1024 + n0 + bn);
        Bsm[(bn+0)*40 + krow] = (__bf16)f.x;
        Bsm[(bn+1)*40 + krow] = (__bf16)f.y;
        Bsm[(bn+2)*40 + krow] = (__bf16)f.z;
        Bsm[(bn+3)*40 + krow] = (__bf16)f.w;
      }
    }
    __syncthreads();
    v16bf a0 = frag_ld(Asm, 40, wm +      (lane & 15), 0, lane);
    v16bf a1 = frag_ld(Asm, 40, wm + 16 + (lane & 15), 0, lane);
    for (int nt = 0; nt < 4; ++nt) {
      v16bf b = frag_ld(Bsm, 40, wn + nt * 16 + (lane & 15), 0, lane);
      acc[0][nt] = wmma_bf16(a0, b, acc[0][nt]);
      acc[1][nt] = wmma_bf16(a1, b, acc[1][nt]);
    }
  }

  const int hi8 = (lane < 16) ? 0 : 8;
  for (int mt = 0; mt < 2; ++mt)
    for (int nt = 0; nt < 4; ++nt) {
      int n = n0 + wn + nt * 16 + (lane & 15);
      float bv = bias[n];
      for (int r = 0; r < 8; ++r) {
        int m = m0 + wm + mt * 16 + r + hi8;
        Out[(size_t)m * 1024 + n] = acc[mt][nt][r] + bv;
      }
    }
}

// ---------------------------------------------------------------------------
extern "C" void kernel_launch(void* const* d_in, const int* in_sizes, int n_in,
                              void* d_out, int out_size, void* d_ws, size_t ws_size,
                              hipStream_t stream) {
  const float* x      = (const float*)d_in[0];
  const float* W_attn = (const float*)d_in[1];
  const float* b_attn = (const float*)d_in[2];
  const float* W_proj = (const float*)d_in[3];
  const float* b_proj = (const float*)d_in[4];
  float* out = (float*)d_out;

  const size_t qkvElems = (size_t)2 * 16 * 2048 * 64;   // B*H*T*D
  __bf16* Qw = (__bf16*)d_ws;
  __bf16* Kw = Qw + qkvElems;
  __bf16* Vw = Kw + qkvElems;
  float*  Yw = (float*)(Vw + qkvElems);                 // 24MB offset, aligned

  qkv_gemm <<<dim3(32, 24), 256, 0, stream>>>(x, W_attn, b_attn, Qw, Kw, Vw);
  attn_kernel<<<dim3(32, 32), 128, 0, stream>>>(Qw, Kw, Vw, Yw);
  proj_gemm <<<dim3(32,  8), 256, 0, stream>>>(Yw, W_proj, b_proj, out);
}
